// SequenceTagger_17497696764699
// MI455X (gfx1250) — compile-verified, matrix operand order checked by reading, not run on hardware
//
#include <hip/hip_runtime.h>
#include <hip/hip_bf16.h>

#define TT   4096
#define DIN  1152
#define HH   512
#define G4H  2048   // 4*H
#define KTAG 24
#define NEGV -10000.0f

typedef __attribute__((ext_vector_type(16))) __bf16 v16bf;
typedef __attribute__((ext_vector_type(8)))  float  v8f;

// ---------- helpers ----------

__device__ __forceinline__ unsigned short f2bf(float f) {
  unsigned int u = __float_as_uint(f);
  unsigned int r = u + 0x7FFFu + ((u >> 16) & 1u);   // round-to-nearest-even
  return (unsigned short)(r >> 16);
}

__device__ __forceinline__ v8f wmma_bf16(v16bf a, v16bf b, v8f c) {
  // 8 args: (neg_a, A, neg_b, B, c_mod, C, reuse_a, reuse_b)
  return __builtin_amdgcn_wmma_f32_16x16x32_bf16(false, a, false, b, (short)0, c,
                                                 false, false);
}

// A/B fragment loader for row-major bf16 matrix (K contiguous, ld multiple of 8).
// ISA 16-bit 16x32 layout: lanes 0-15 hold row (lane), K = k0+0..7 and k0+16..23;
// lanes 16-31 hold row (lane-16), K = k0+8..15 and k0+24..31.
__device__ __forceinline__ v16bf load_frag(const unsigned short* __restrict__ base,
                                           int ld, int row, int k0, int lane) {
  int koff = k0 + ((lane & 16) ? 8 : 0);
  const unsigned short* p = base + (size_t)row * ld + koff;
  union { v16bf v; uint4 q[2]; } f;
  f.q[0] = *reinterpret_cast<const uint4*>(p);
  f.q[1] = *reinterpret_cast<const uint4*>(p + 16);
  return f.v;
}

// B fragment where every column n holds the same vector h (for matvec via WMMA):
// B[kk, n] = h[k0+kk] for all 16 columns.
__device__ __forceinline__ v16bf load_frag_h(const unsigned int* hw, int k0, int lane) {
  int koff = k0 + ((lane & 16) ? 8 : 0);
  union { v16bf v; unsigned int u[8]; } f;
#pragma unroll
  for (int i = 0; i < 4; ++i) {
    f.u[i]     = hw[(koff >> 1) + i];
    f.u[4 + i] = hw[((koff + 16) >> 1) + i];
  }
  return f.v;
}

// ---------- kernels ----------

__global__ void cvt_f32_bf16(const float* __restrict__ src,
                             unsigned short* __restrict__ dst, int n) {
  int i = blockIdx.x * blockDim.x + threadIdx.x;
  int stride = gridDim.x * blockDim.x;
  for (; i < n; i += stride) dst[i] = f2bf(src[i]);
}

// C[M,Nvalid] = A[M,K] * B[Nvalid,K]^T (+bias). Npad = N rounded up to 64.
// One wave computes a 16x64 tile (A fragment reused across 4 WMMAs).
__global__ void gemm_bf16(const unsigned short* __restrict__ A,
                          const unsigned short* __restrict__ B,
                          float* __restrict__ C, const float* __restrict__ bias,
                          int M, int K, int Npad, int Nvalid) {
  const int lane = threadIdx.x & 31;
  const int gw = blockIdx.x * (blockDim.x >> 5) + (threadIdx.x >> 5);
  const int ntiles = Npad >> 6;
  const int m0 = (gw / ntiles) << 4;
  const int n0 = (gw % ntiles) << 6;
  if (m0 >= M) return;

  v8f acc0 = {}, acc1 = {}, acc2 = {}, acc3 = {};
  const int arow = m0 + (lane & 15);
  const int br = n0 + (lane & 15);
  const int b0r = (br      < Nvalid) ? br      : Nvalid - 1;
  const int b1r = (br + 16 < Nvalid) ? br + 16 : Nvalid - 1;
  const int b2r = (br + 32 < Nvalid) ? br + 32 : Nvalid - 1;
  const int b3r = (br + 48 < Nvalid) ? br + 48 : Nvalid - 1;

  for (int k0 = 0; k0 < K; k0 += 32) {
    v16bf a  = load_frag(A, K, arow, k0, lane);
    v16bf b0 = load_frag(B, K, b0r, k0, lane);
    v16bf b1 = load_frag(B, K, b1r, k0, lane);
    v16bf b2 = load_frag(B, K, b2r, k0, lane);
    v16bf b3 = load_frag(B, K, b3r, k0, lane);
    acc0 = wmma_bf16(a, b0, acc0);
    acc1 = wmma_bf16(a, b1, acc1);
    acc2 = wmma_bf16(a, b2, acc2);
    acc3 = wmma_bf16(a, b3, acc3);
  }

  const int cmBase = m0 + ((lane & 16) ? 8 : 0);
  const int cnBase = n0 + (lane & 15);
  v8f accs[4] = {acc0, acc1, acc2, acc3};
#pragma unroll
  for (int j = 0; j < 4; ++j) {
    int cn = cnBase + j * 16;
    if (cn >= Nvalid) continue;
    float bv = bias ? bias[cn] : 0.0f;
#pragma unroll
    for (int r = 0; r < 8; ++r)
      C[(size_t)(cmBase + r) * Nvalid + cn] = accs[j][r] + bv;
  }
}

// One block per direction (blockIdx.x: 0=fwd, 1=bwd), 512 threads = 16 waves.
// Per step: z_hh = Whh @ h via WMMA (A = Whh row tile, B = h broadcast columns),
// then gate math per thread (thread tid owns hidden index tid; c stays in reg).
__global__ void __launch_bounds__(512)
lstm_bidir(const float* __restrict__ z_f, const float* __restrict__ z_b,
           const unsigned short* __restrict__ whh_f,
           const unsigned short* __restrict__ whh_b,
           const float* __restrict__ bias_f, const float* __restrict__ bias_b,
           const float* __restrict__ h0f, const float* __restrict__ c0f,
           const float* __restrict__ h0b, const float* __restrict__ c0b,
           unsigned short* __restrict__ y_bf /* [T, 2H] bf16 */) {
  const int dir = blockIdx.x;
  const float* z = dir ? z_b : z_f;
  const unsigned short* whh = dir ? whh_b : whh_f;
  const float* bias = dir ? bias_b : bias_f;
  const float* h0r = dir ? h0b : h0f;
  const float* c0r = dir ? c0b : c0f;

  __shared__ unsigned short h_sh[HH];
  __shared__ float zhh[G4H];
  const unsigned int* hw = reinterpret_cast<const unsigned int*>(h_sh);

  const int tid = threadIdx.x;        // 0..511
  const int lane = tid & 31;
  const int wave = tid >> 5;          // 0..15

  float c_priv = c0r[tid];
  h_sh[tid] = f2bf(h0r[tid]);
  const float bi = bias[tid], bff = bias[HH + tid],
              bg = bias[2 * HH + tid], bo = bias[3 * HH + tid];
  __syncthreads();

  for (int step = 0; step < TT; ++step) {
    const int t = dir ? (TT - 1 - step) : step;
    // ---- z_hh = Whh @ h : 128 tiles of 16 outputs, 8 tiles per wave ----
#pragma unroll 2
    for (int tile = 0; tile < 8; ++tile) {
      const int j0 = (wave * 8 + tile) * 16;
      v8f acc = {};
#pragma unroll
      for (int kc = 0; kc < 16; ++kc) {
        v16bf a = load_frag(whh, HH, j0 + (lane & 15), kc * 32, lane);
        v16bf b = load_frag_h(hw, kc * 32, lane);
        acc = wmma_bf16(a, b, acc);
      }
      if ((lane & 15) == 0) {         // lanes 0 (M=0..7) and 16 (M=8..15), N=0
        const int base = j0 + ((lane & 16) ? 8 : 0);
#pragma unroll
        for (int r = 0; r < 8; ++r) zhh[base + r] = acc[r];
      }
    }
    __syncthreads();

    const float* zr = z + (size_t)t * G4H;
    float gi = zr[tid]          + zhh[tid]          + bi;
    float gf = zr[HH + tid]     + zhh[HH + tid]     + bff;
    float gg = zr[2 * HH + tid] + zhh[2 * HH + tid] + bg;
    float go = zr[3 * HH + tid] + zhh[3 * HH + tid] + bo;
    float si = 1.0f / (1.0f + __expf(-gi));
    float sf = 1.0f / (1.0f + __expf(-gf));
    float so = 1.0f / (1.0f + __expf(-go));
    c_priv = sf * c_priv + si * tanhf(gg);
    float h = so * tanhf(c_priv);
    __syncthreads();                  // all zhh/h_sh readers done
    unsigned short hb = f2bf(h);
    h_sh[tid] = hb;
    y_bf[(size_t)t * (2 * HH) + dir * HH + tid] = hb;
    __syncthreads();
  }
}

// Single wave32 Viterbi: lane k owns tag k. bps stored as bytes in scratch.
__global__ void viterbi_kernel(const float* __restrict__ feats,
                               const float* __restrict__ trans,
                               float* __restrict__ out,
                               unsigned char* __restrict__ bps) {
  const int lane = threadIdx.x;
  float trow[KTAG];
#pragma unroll
  for (int j = 0; j < KTAG; ++j)
    trow[j] = (lane < KTAG) ? trans[lane * KTAG + j] : NEGV;
  float fv = (lane == 0) ? 0.0f : NEGV;      // START = 0

  for (int t = 0; t < TT; ++t) {
    float best = -3.0e38f; int bj = 0;
#pragma unroll
    for (int j = 0; j < KTAG; ++j) {
      float v = __shfl(fv, j, 32) + trow[j];
      if (v > best) { best = v; bj = j; }
    }
    float ft = (lane < KTAG) ? feats[(size_t)t * KTAG + lane] : NEGV;
    fv = best + ft;
    if (lane < KTAG) bps[(size_t)t * KTAG + lane] = (unsigned char)bj;
  }

  float term = fv + ((lane < KTAG) ? trans[1 * KTAG + lane] : NEGV); // STOP row
  if (lane <= 1) term = NEGV;                 // kill START(0) and STOP(1)
  if (lane >= KTAG) term = -3.0e38f;
  float bestv = -3.0e38f; int besti = 0;
#pragma unroll
  for (int j = 0; j < KTAG; ++j) {
    float tj = __shfl(term, j, 32);
    if (tj > bestv) { bestv = tj; besti = j; }
  }
  if (lane == 0) {
    out[0] = bestv;
    int tag = besti;
    out[1 + (TT - 1)] = (float)tag;
    for (int t = TT - 1; t >= 1; --t) {
      tag = bps[(size_t)t * KTAG + tag];
      out[1 + (t - 1)] = (float)tag;
    }
  }
}

// ---------- host ----------

extern "C" void kernel_launch(void* const* d_in, const int* in_sizes, int n_in,
                              void* d_out, int out_size, void* d_ws, size_t ws_size,
                              hipStream_t stream) {
  const float* sentence = (const float*)d_in[0];
  const float* Wih0f = (const float*)d_in[1];
  const float* Whh0f = (const float*)d_in[2];
  const float* b0f   = (const float*)d_in[3];
  const float* Wih0b = (const float*)d_in[4];
  const float* Whh0b = (const float*)d_in[5];
  const float* b0b   = (const float*)d_in[6];
  const float* Wih1f = (const float*)d_in[7];
  const float* Whh1f = (const float*)d_in[8];
  const float* b1f   = (const float*)d_in[9];
  const float* Wih1b = (const float*)d_in[10];
  const float* Whh1b = (const float*)d_in[11];
  const float* b1b   = (const float*)d_in[12];
  const float* W_tag = (const float*)d_in[13];
  const float* b_tag = (const float*)d_in[14];
  const float* h0    = (const float*)d_in[15];
  const float* c0    = (const float*)d_in[16];
  const float* trans = (const float*)d_in[17];
  float* out = (float*)d_out;

  char* ws = (char*)d_ws;
  size_t off = 0;
  auto alloc = [&](size_t bytes) -> char* {
    char* p = ws + off;
    off = (off + bytes + 255) & ~(size_t)255;
    return p;
  };

  unsigned short* sbf     = (unsigned short*)alloc((size_t)TT * DIN * 2);
  unsigned short* wih0fB  = (unsigned short*)alloc((size_t)G4H * DIN * 2);
  unsigned short* wih0bB  = (unsigned short*)alloc((size_t)G4H * DIN * 2);
  unsigned short* wih1fB  = (unsigned short*)alloc((size_t)G4H * 2 * HH * 2);
  unsigned short* wih1bB  = (unsigned short*)alloc((size_t)G4H * 2 * HH * 2);
  unsigned short* whh0fB  = (unsigned short*)alloc((size_t)G4H * HH * 2);
  unsigned short* whh0bB  = (unsigned short*)alloc((size_t)G4H * HH * 2);
  unsigned short* whh1fB  = (unsigned short*)alloc((size_t)G4H * HH * 2);
  unsigned short* whh1bB  = (unsigned short*)alloc((size_t)G4H * HH * 2);
  unsigned short* wtagB   = (unsigned short*)alloc((size_t)KTAG * 2 * HH * 2);
  float*          zA      = (float*)alloc((size_t)TT * G4H * 4);   // fwd Z, reused by layer1
  float*          zB      = (float*)alloc((size_t)TT * G4H * 4);   // bwd Z, reused by layer1
  unsigned short* y0bf    = (unsigned short*)alloc((size_t)TT * 2 * HH * 2);
  unsigned short* y1bf    = (unsigned short*)alloc((size_t)TT * 2 * HH * 2);
  float*          feats   = (float*)alloc((size_t)TT * KTAG * 4);
  unsigned char*  bps     = (unsigned char*)alloc((size_t)TT * KTAG);
  (void)ws_size; (void)in_sizes; (void)n_in; (void)out_size;

  auto cvt = [&](const float* s, unsigned short* d, int n) {
    int blocks = (n + 2047) / 2048;
    cvt_f32_bf16<<<dim3(blocks), dim3(256), 0, stream>>>(s, d, n);
  };
  cvt(sentence, sbf, TT * DIN);
  cvt(Wih0f, wih0fB, G4H * DIN);
  cvt(Wih0b, wih0bB, G4H * DIN);
  cvt(Wih1f, wih1fB, G4H * 2 * HH);
  cvt(Wih1b, wih1bB, G4H * 2 * HH);
  cvt(Whh0f, whh0fB, G4H * HH);
  cvt(Whh0b, whh0bB, G4H * HH);
  cvt(Whh1f, whh1fB, G4H * HH);
  cvt(Whh1b, whh1bB, G4H * HH);
  cvt(W_tag, wtagB, KTAG * 2 * HH);

  // ---- layer 0 input GEMMs: Z = x @ Wih^T   [4096 x 2048] ----
  {
    int waves = (TT / 16) * (G4H / 64);        // 8192
    gemm_bf16<<<dim3(waves / 8), dim3(256), 0, stream>>>(
        sbf, wih0fB, zA, nullptr, TT, DIN, G4H, G4H);
    gemm_bf16<<<dim3(waves / 8), dim3(256), 0, stream>>>(
        sbf, wih0bB, zB, nullptr, TT, DIN, G4H, G4H);
  }
  // ---- layer 0 recurrence (both directions concurrently) ----
  lstm_bidir<<<dim3(2), dim3(512), 0, stream>>>(
      zA, zB, whh0fB, whh0bB, b0f, b0b,
      h0 + 0 * HH, c0 + 0 * HH, h0 + 1 * HH, c0 + 1 * HH, y0bf);

  // ---- layer 1 input GEMMs (Z buffers reused) ----
  {
    int waves = (TT / 16) * (G4H / 64);
    gemm_bf16<<<dim3(waves / 8), dim3(256), 0, stream>>>(
        y0bf, wih1fB, zA, nullptr, TT, 2 * HH, G4H, G4H);
    gemm_bf16<<<dim3(waves / 8), dim3(256), 0, stream>>>(
        y0bf, wih1bB, zB, nullptr, TT, 2 * HH, G4H, G4H);
  }
  // ---- layer 1 recurrence ----
  lstm_bidir<<<dim3(2), dim3(512), 0, stream>>>(
      zA, zB, whh1fB, whh1bB, b1f, b1b,
      h0 + 2 * HH, c0 + 2 * HH, h0 + 3 * HH, c0 + 3 * HH, y1bf);

  // ---- tag projection: feats = y1 @ W_tag^T + b_tag  [4096 x 24] ----
  {
    int waves = (TT / 16) * (64 / 64);         // 256
    gemm_bf16<<<dim3(waves / 8), dim3(256), 0, stream>>>(
        y1bf, wtagB, feats, b_tag, TT, 2 * HH, 64, KTAG);
  }
  // ---- Viterbi decode + backtrack ----
  viterbi_kernel<<<dim3(1), dim3(32), 0, stream>>>(feats, trans, out, bps);
}
